// GCNModel_89893665506085
// MI455X (gfx1250) — matrix operand order repaired and checked by reading, not compiled, not run on hardware
//
#include <hip/hip_runtime.h>

typedef __attribute__((ext_vector_type(16))) _Float16 v16h;
typedef __attribute__((ext_vector_type(8)))  float    v8f;

static __device__ __forceinline__ void fatomic_add(float* p, float v) {
    (void)__hip_atomic_fetch_add(p, v, __ATOMIC_RELAXED, __HIP_MEMORY_SCOPE_AGENT);
}

// ---- degree: count incoming edges (self-loop added analytically later) ----
__global__ void k_degree(const int* __restrict__ col, int E, unsigned* __restrict__ deg) {
    int e = blockIdx.x * blockDim.x + threadIdx.x;
    if (e < E) atomicAdd(&deg[col[e]], 1u);
}

// ---- dinv = rsqrt(deg + 1)  (self-loop makes deg >= 1, no zero branch) ----
__global__ void k_dinv(const unsigned* __restrict__ deg, float* __restrict__ dinv, int n) {
    int i = blockIdx.x * blockDim.x + threadIdx.x;
    if (i < n) dinv[i] = rsqrtf((float)(deg[i] + 1u));
}

// ---- layer-1 GEMM: h[N,64] = x[N,2] @ W1[2,64] via WMMA f32<-f16, K padded 2->32 ----
__global__ void k_gemm1(const float* __restrict__ x, const float* __restrict__ W1,
                        float* __restrict__ h, int n, int ntiles) {
    int wave = (int)((blockIdx.x * blockDim.x + threadIdx.x) >> 5);
    int lane = threadIdx.x & 31;
    if (wave >= ntiles) return;               // wave-uniform: EXEC all-1s for WMMA
    int mbase = wave * 16;
    bool full = (mbase + 16 <= n);            // wave-uniform fast path (true for N=100000)

    // A: 16x32 f16. Lanes 0-15 carry K=0..7 in elems 0..7; only K=0,1 nonzero.
    v16h a;
#pragma unroll
    for (int i = 0; i < 16; ++i) a[i] = (_Float16)0.0f;
    int mr = mbase + (lane & 15);
    if (lane < 16 && mr < n) {
        a[0] = (_Float16)x[2 * mr + 0];
        a[1] = (_Float16)x[2 * mr + 1];
    }

    int mo = mbase + (lane >= 16 ? 8 : 0);
#pragma unroll
    for (int blk = 0; blk < 4; ++blk) {       // N=64 in four 16-wide column blocks
        // B: 32x16 f16. Lanes 0-15 carry K=0..15 (elems 0..15); only K=0,1 nonzero.
        v16h b;
#pragma unroll
        for (int i = 0; i < 16; ++i) b[i] = (_Float16)0.0f;
        if (lane < 16) {
            int nn = blk * 16 + lane;
            b[0] = (_Float16)W1[nn];          // W1[0][nn]
            b[1] = (_Float16)W1[64 + nn];     // W1[1][nn]
        }
        v8f c;
#pragma unroll
        for (int i = 0; i < 8; ++i) c[i] = 0.0f;
        c = __builtin_amdgcn_wmma_f32_16x16x32_f16(false, a, false, b, (short)0, c, false, false);

        // D: VGPR r, lanes 0-15 -> M=r, lanes 16-31 -> M=r+8; N = (lane&15)+16*blk
        int nc = blk * 16 + (lane & 15);
        float* hp = h + (size_t)mo * 64 + nc;
        if (full) {
#pragma unroll
            for (int r = 0; r < 8; ++r) hp[(size_t)r * 64] = c[r];
        } else {
#pragma unroll
            for (int r = 0; r < 8; ++r)
                if (mo + r < n) hp[(size_t)r * 64] = c[r];
        }
    }
}

// ---- layer-1 edge scatter: agg[col] += h[row]*norm, 16 threads/edge, float4 each ----
__global__ void k_scatter1(const int* __restrict__ ei, int E,
                           const float* __restrict__ dinv,
                           const float* __restrict__ h,
                           float* __restrict__ agg) {
    long long t = (long long)blockIdx.x * blockDim.x + threadIdx.x;
    int e = (int)(t >> 4);
    if (e >= E) return;
    int f = ((int)t & 15) * 4;
    int r = ei[e];
    int c = ei[(size_t)E + e];
    float norm = dinv[r] * dinv[c];
    float4 hv = *(const float4*)(h + (size_t)r * 64 + f);
    float* ap = agg + (size_t)c * 64 + f;
    fatomic_add(ap + 0, hv.x * norm);
    fatomic_add(ap + 1, hv.y * norm);
    fatomic_add(ap + 2, hv.z * norm);
    fatomic_add(ap + 3, hv.w * norm);
}

// ---- layer-2 GEMM fused with (self-loop + bias + ReLU):
//      h2[m] = relu(agg[m]+h[m]*dinv[m]^2+b1) . W2   via two chained WMMAs (K=64) ----
__global__ void k_gemm2(const float* __restrict__ agg, const float* __restrict__ h,
                        const float* __restrict__ dinv, const float* __restrict__ b1,
                        const float* __restrict__ W2, float* __restrict__ h2,
                        int n, int ntiles) {
    int wave = (int)((blockIdx.x * blockDim.x + threadIdx.x) >> 5);
    int lane = threadIdx.x & 31;
    if (wave >= ntiles) return;               // wave-uniform guard
    int mbase = wave * 16;
    bool full = (mbase + 16 <= n);            // wave-uniform fast path
    int m = mbase + (lane & 15);
    bool valid = full || (m < n);
    float d2 = valid ? dinv[m] * dinv[m] : 0.0f;
    int kh = (lane >> 4) * 8;                 // A: lane-half K interleave offset
    int kh2 = (lane >> 4) * 16;               // (unused)

    const float* am = agg + (size_t)m * 64;
    const float* hm = h + (size_t)m * 64;

    v8f c;
#pragma unroll
    for (int i = 0; i < 8; ++i) c[i] = 0.0f;

#pragma unroll
    for (int kc = 0; kc < 2; ++kc) {
        int kbase = kc * 32;
        // A: 16x32 f16 chunk of relu'd node features.
        // lanes 0-15: elems 0..7 -> K=kbase+0..7, elems 8..15 -> K=kbase+16..23
        // lanes 16-31: +8 on both ranges.
        v16h a;
        if (full) {
#pragma unroll
            for (int i = 0; i < 8; ++i) {
                int K1 = kbase + kh + i;
                int K2 = kbase + 16 + kh + i;
                float v1 = am[K1] + hm[K1] * d2 + b1[K1];
                float v2 = am[K2] + hm[K2] * d2 + b1[K2];
                a[i]     = (_Float16)(v1 > 0.0f ? v1 : 0.0f);
                a[i + 8] = (_Float16)(v2 > 0.0f ? v2 : 0.0f);
            }
        } else {
#pragma unroll
            for (int i = 0; i < 16; ++i) a[i] = (_Float16)0.0f;
            if (valid) {
#pragma unroll
                for (int i = 0; i < 8; ++i) {
                    int K1 = kbase + kh + i;
                    int K2 = kbase + 16 + kh + i;
                    float v1 = am[K1] + hm[K1] * d2 + b1[K1];
                    float v2 = am[K2] + hm[K2] * d2 + b1[K2];
                    a[i]     = (_Float16)(v1 > 0.0f ? v1 : 0.0f);
                    a[i + 8] = (_Float16)(v2 > 0.0f ? v2 : 0.0f);
                }
            }
        }
        // B: 32x16 f16, only column N=0 nonzero (W2 is 64x1).
        // lanes with (lane&15)==0 carry N=0, same K interleave as A (elems 0..7 -> K=kbase+kh+0..7, 8..15 -> K=kbase+16+kh+0..7)
        v16h b;
#pragma unroll
        for (int i = 0; i < 16; ++i) b[i] = (_Float16)0.0f;
        if ((lane & 15) == 0) {
#pragma unroll
            for (int i = 0; i < 16; ++i) b[i] = (_Float16)W2[kbase + kh + i + (i >= 8 ? 8 : 0)];
        }
        c = __builtin_amdgcn_wmma_f32_16x16x32_f16(false, a, false, b, (short)0, c, false, false);
    }

    // D column N=0 lives in lanes 0 (M=r) and 16 (M=r+8)
    if ((lane & 15) == 0) {
        int mo = mbase + (lane >= 16 ? 8 : 0);
        if (full) {
#pragma unroll
            for (int r = 0; r < 8; ++r) h2[mo + r] = c[r];
        } else {
#pragma unroll
            for (int r = 0; r < 8; ++r)
                if (mo + r < n) h2[mo + r] = c[r];
        }
    }
}

// ---- layer-2 edge scatter: out[col] += h2[row]*norm ----
__global__ void k_scatter2(const int* __restrict__ ei, int E,
                           const float* __restrict__ dinv,
                           const float* __restrict__ h2, float* __restrict__ out) {
    int e = blockIdx.x * blockDim.x + threadIdx.x;
    if (e >= E) return;
    int r = ei[e];
    int c = ei[(size_t)E + e];
    fatomic_add(&out[c], h2[r] * dinv[r] * dinv[c]);
}

// ---- self-loop + bias for layer 2 ----
__global__ void k_finalize(const float* __restrict__ h2, const float* __restrict__ dinv,
                           const float* __restrict__ b2, float* __restrict__ out, int n) {
    int i = blockIdx.x * blockDim.x + threadIdx.x;
    if (i < n) out[i] += h2[i] * dinv[i] * dinv[i] + b2[0];
}

extern "C" void kernel_launch(void* const* d_in, const int* in_sizes, int n_in,
                              void* d_out, int out_size, void* d_ws, size_t ws_size,
                              hipStream_t stream) {
    const float* x  = (const float*)d_in[0];
    const int*   ei = (const int*)d_in[1];
    const float* W1 = (const float*)d_in[2];
    const float* b1 = (const float*)d_in[3];
    const float* W2 = (const float*)d_in[4];
    const float* b2 = (const float*)d_in[5];
    float* out = (float*)d_out;

    const int n = in_sizes[0] / 2;   // 100000
    const int E = in_sizes[1] / 2;   // 3200000

    auto al = [](size_t v) { return (v + 255) & ~(size_t)255; };
    char* w = (char*)d_ws;
    size_t off = 0;
    unsigned* deg  = (unsigned*)(w + off); off = al(off + (size_t)n * 4);
    float*    dinv = (float*)(w + off);    off = al(off + (size_t)n * 4);
    float*    h    = (float*)(w + off);    off = al(off + (size_t)n * 64 * 4);
    float*    agg  = (float*)(w + off);    off = al(off + (size_t)n * 64 * 4);
    float*    h2   = (float*)(w + off);    off = al(off + (size_t)n * 4);

    hipMemsetAsync(deg, 0, (size_t)n * 4, stream);
    hipMemsetAsync(agg, 0, (size_t)n * 64 * 4, stream);
    hipMemsetAsync(out, 0, (size_t)n * 4, stream);

    const int ntiles = (n + 15) / 16;
    dim3 B(256);

    k_degree<<<(E + 255) / 256, B, 0, stream>>>(ei + E, E, deg);
    k_dinv<<<(n + 255) / 256, B, 0, stream>>>(deg, dinv, n);
    k_gemm1<<<(ntiles + 7) / 8, B, 0, stream>>>(x, W1, h, n, ntiles);
    long long sc1 = (long long)E * 16;
    k_scatter1<<<(unsigned)((sc1 + 255) / 256), B, 0, stream>>>(ei, E, dinv, h, agg);
    k_gemm2<<<(ntiles + 7) / 8, B, 0, stream>>>(agg, h, dinv, b1, W2, h2, n, ntiles);
    k_scatter2<<<(E + 255) / 256, B, 0, stream>>>(ei, E, dinv, h2, out);
    k_finalize<<<(n + 255) / 256, B, 0, stream>>>(h2, dinv, b2, out, n);
}
